// GraphAggregator_8065948582552
// MI455X (gfx1250) — compile-verified, matrix operand order checked by reading, not run on hardware
//
#include <hip/hip_runtime.h>

typedef __attribute__((ext_vector_type(16))) _Float16 v16h;
typedef __attribute__((ext_vector_type(8)))  _Float16 v8h;
typedef __attribute__((ext_vector_type(4)))  _Float16 v4h;
typedef __attribute__((ext_vector_type(8)))  float    v8f;
typedef __attribute__((ext_vector_type(4)))  float    v4f;

#define N_NODES   100000
#define FEAT      128
#define NSAMP     10
#define LDSTRIDE  136     // halfs per LDS row: 128 + 8 pad -> spreads banks
#define OSTR      20      // floats per row of out-transpose tile: 16 + 4 pad
#define TILES     ((N_NODES + 15) / 16)   // 6250, exact

// Build a 16-half WMMA fragment from two contiguous 16B LDS chunks
// (A-matrix 16-bit layout: lane holds K = half*8..+7 and K = 16+half*8..+7).
static __device__ __forceinline__ v16h frag16(const _Float16* p0) {
  v8h lo = *(const v8h*)(p0);
  v8h hi = *(const v8h*)(p0 + 16);
  return __builtin_shufflevector(lo, hi, 0,1,2,3,4,5,6,7,8,9,10,11,12,13,14,15);
}

static __device__ __forceinline__ v4f cvt4(v4h h) {
  v4f r; r.x = (float)h.x; r.y = (float)h.y; r.z = (float)h.z; r.w = (float)h.w;
  return r;
}

// ---------------- Layer 1: h1 = relu(mean({self}+adj) @ W1^T), all nodes ----
__global__ __launch_bounds__(256) void sage1(const float* __restrict__ feat,
                                             const float* __restrict__ W1,
                                             const int*   __restrict__ adj,
                                             _Float16*    __restrict__ h1)
{
  __shared__ __align__(16) _Float16 sW[128 * LDSTRIDE];
  __shared__ __align__(16) _Float16 sA[8][16 * LDSTRIDE];

  const int tid  = threadIdx.x;
  const int lane = tid & 31;
  const int wave = tid >> 5;

  // Stage W1 (f32 -> f16) into padded LDS; B-fragment of W1^T == A-layout of W1.
  for (int i = tid; i < 128 * 128; i += 256)
    sW[(i >> 7) * LDSTRIDE + (i & 127)] = (_Float16)W1[i];
  __syncthreads();

  const int tile = blockIdx.x * 8 + wave;           // 16 nodes per wave
  if (tile < TILES) {
    const int base = tile * 16;
    _Float16* at = &sA[wave][0];
    const float inv = 1.0f / (float)(NSAMP + 1);

    // Gather + mean: each wave reads full 512B rows (float4/lane), f32 accum.
    for (int r = 0; r < 16; ++r) {
      const int node = base + r;
      v4f acc = *(const v4f*)(feat + (long)node * FEAT + lane * 4);
      #pragma unroll
      for (int s = 0; s < NSAMP; ++s) {
        const int nb = adj[node * NSAMP + s];
        acc += *(const v4f*)(feat + (long)nb * FEAT + lane * 4);
      }
      acc *= inv;
      v4h hv;
      hv.x = (_Float16)acc.x; hv.y = (_Float16)acc.y;
      hv.z = (_Float16)acc.z; hv.w = (_Float16)acc.w;
      *(v4h*)(at + r * LDSTRIDE + lane * 4) = hv;
    }
    asm volatile("s_wait_dscnt 0" ::: "memory");     // tile visible to all lanes

    const int m = lane & 15, half = lane >> 4;
    v16h a[4];
    #pragma unroll
    for (int kc = 0; kc < 4; ++kc)
      a[kc] = frag16(at + m * LDSTRIDE + kc * 32 + half * 8);
    asm volatile("s_wait_dscnt 0" ::: "memory");     // frags in VGPRs; tile reusable

    #pragma unroll
    for (int nt = 0; nt < 8; ++nt) {                 // 8 output N-tiles of 16
      v8f c = {};
      #pragma unroll
      for (int kc = 0; kc < 4; ++kc) {               // K = 128 in 4 chunks of 32
        v16h b = frag16(sW + (nt * 16 + m) * LDSTRIDE + kc * 32 + half * 8);
        c = __builtin_amdgcn_wmma_f32_16x16x32_f16(false, a[kc], false, b,
                                                   (short)0, c, false, false);
      }
      // relu, repack into LDS tile (row-major) for coalesced global store
      #pragma unroll
      for (int i = 0; i < 8; ++i) {
        float v = c[i] > 0.0f ? c[i] : 0.0f;
        at[(i + 8 * half) * LDSTRIDE + nt * 16 + m] = (_Float16)v;
      }
    }
    asm volatile("s_wait_dscnt 0" ::: "memory");

    for (int r = 0; r < 16; ++r) {                   // 256B/row coalesced f16 store
      v4h hv = *(const v4h*)(at + r * LDSTRIDE + lane * 4);
      *(v4h*)(h1 + (long)(base + r) * FEAT + lane * 4) = hv;
    }
  }
}

// ---- Layer 2: out = relu(mean(h1[{nodes}+adj[nodes]]) @ W2^T)^T  [128,N] ----
__global__ __launch_bounds__(256) void sage2(const _Float16* __restrict__ h1,
                                             const float*    __restrict__ W2,
                                             const int*      __restrict__ adj,
                                             const int*      __restrict__ nodes,
                                             float*          __restrict__ out)
{
  __shared__ __align__(16) _Float16 sW[128 * LDSTRIDE];
  __shared__ __align__(16) _Float16 sA[8][16 * LDSTRIDE];
  __shared__ __align__(16) float    sO[8][128 * OSTR];   // [embed e][batch b] tile

  const int tid  = threadIdx.x;
  const int lane = tid & 31;
  const int wave = tid >> 5;

  for (int i = tid; i < 128 * 128; i += 256)
    sW[(i >> 7) * LDSTRIDE + (i & 127)] = (_Float16)W2[i];
  __syncthreads();

  const int tile = blockIdx.x * 8 + wave;
  if (tile < TILES) {
    const int base = tile * 16;
    _Float16* at = &sA[wave][0];
    float*    ot = &sO[wave][0];
    const float inv = 1.0f / (float)(NSAMP + 1);

    for (int r = 0; r < 16; ++r) {
      const int self = nodes[base + r];
      v4f acc = cvt4(*(const v4h*)(h1 + (long)self * FEAT + lane * 4));
      #pragma unroll
      for (int s = 0; s < NSAMP; ++s) {
        const int nb = adj[self * NSAMP + s];
        acc += cvt4(*(const v4h*)(h1 + (long)nb * FEAT + lane * 4));
      }
      acc *= inv;
      v4h hv;
      hv.x = (_Float16)acc.x; hv.y = (_Float16)acc.y;
      hv.z = (_Float16)acc.z; hv.w = (_Float16)acc.w;
      *(v4h*)(at + r * LDSTRIDE + lane * 4) = hv;
    }
    asm volatile("s_wait_dscnt 0" ::: "memory");

    const int m = lane & 15, half = lane >> 4;
    v16h a[4];
    #pragma unroll
    for (int kc = 0; kc < 4; ++kc)
      a[kc] = frag16(at + m * LDSTRIDE + kc * 32 + half * 8);

    #pragma unroll
    for (int nt = 0; nt < 8; ++nt) {
      v8f c = {};
      #pragma unroll
      for (int kc = 0; kc < 4; ++kc) {
        v16h b = frag16(sW + (nt * 16 + m) * LDSTRIDE + kc * 32 + half * 8);
        c = __builtin_amdgcn_wmma_f32_16x16x32_f16(false, a[kc], false, b,
                                                   (short)0, c, false, false);
      }
      // relu + transpose through LDS: out tile stored [e][b], padded stride
      #pragma unroll
      for (int i = 0; i < 8; ++i) {
        float v = c[i] > 0.0f ? c[i] : 0.0f;
        ot[(nt * 16 + m) * OSTR + i + 8 * half] = v;
      }
    }
    asm volatile("s_wait_dscnt 0" ::: "memory");

    // Coalesced stores: per e-row 16 consecutive b -> 64B segments, float4/lane
    #pragma unroll
    for (int ii = 0; ii < 16; ++ii) {
      const int e  = ii * 8 + (lane >> 2);
      const int b4 = (lane & 3) * 4;
      v4f v = *(const v4f*)(ot + e * OSTR + b4);
      *(v4f*)(out + (long)e * N_NODES + base + b4) = v;
    }
  }
}

extern "C" void kernel_launch(void* const* d_in, const int* in_sizes, int n_in,
                              void* d_out, int out_size, void* d_ws, size_t ws_size,
                              hipStream_t stream)
{
  const float* feat  = (const float*)d_in[0];   // [100000,128] f32
  const float* W1    = (const float*)d_in[1];   // [128,128] f32
  const float* W2    = (const float*)d_in[2];   // [128,128] f32
  const int*   adj   = (const int*)  d_in[3];   // [100000,10] int
  const int*   nodes = (const int*)  d_in[4];   // [100000] int
  float*       out   = (float*)d_out;           // [128,100000] f32
  _Float16*    h1    = (_Float16*)d_ws;         // 100000*128 f16 = 25.6 MB scratch

  const int blocks = (TILES + 7) / 8;           // 8 waves (tiles) per 256-thr block
  sage1<<<blocks, 256, 0, stream>>>(feat, W1, adj, h1);
  sage2<<<blocks, 256, 0, stream>>>(h1, W2, adj, nodes, out);
}
